// AttentionCombinerWithResidual_24223615550547
// MI455X (gfx1250) — matrix operand order repaired and checked by reading, not compiled, not run on hardware
//
#include <hip/hip_runtime.h>

typedef __bf16 bf16;
typedef __attribute__((ext_vector_type(4)))  bf16 bf16x4;
typedef __attribute__((ext_vector_type(8)))  bf16 bf16x8;
typedef __attribute__((ext_vector_type(16))) bf16 bf16x16;
typedef __attribute__((ext_vector_type(8)))  float v8f;

#define DMODEL 256
#define EPS 1e-5f

// ---------- wave helpers ----------

// reduce across a 16-lane group (xor masks 1,2,4,8 stay within the group)
__device__ __forceinline__ float red_max16(float x) {
  x = fmaxf(x, __int_as_float(__builtin_amdgcn_ds_swizzle(__float_as_int(x), 0x041F)));
  x = fmaxf(x, __int_as_float(__builtin_amdgcn_ds_swizzle(__float_as_int(x), 0x081F)));
  x = fmaxf(x, __int_as_float(__builtin_amdgcn_ds_swizzle(__float_as_int(x), 0x101F)));
  x = fmaxf(x, __int_as_float(__builtin_amdgcn_ds_swizzle(__float_as_int(x), 0x201F)));
  return x;
}
__device__ __forceinline__ float red_sum16(float x) {
  x += __int_as_float(__builtin_amdgcn_ds_swizzle(__float_as_int(x), 0x041F));
  x += __int_as_float(__builtin_amdgcn_ds_swizzle(__float_as_int(x), 0x081F));
  x += __int_as_float(__builtin_amdgcn_ds_swizzle(__float_as_int(x), 0x101F));
  x += __int_as_float(__builtin_amdgcn_ds_swizzle(__float_as_int(x), 0x201F));
  return x;
}

// A operand (16x32 bf16, MxK): lane (m = lane&15, h = lane>>4) holds row m,
// VGPR0-3 = K[8h .. 8h+7], VGPR4-7 = K[16+8h .. 16+8h+7]  (per ISA 7.12.2)
__device__ __forceinline__ bf16x16 load_a(const bf16* base, int ld) {
  int lane = threadIdx.x & 31;
  int m = lane & 15, h = lane >> 4;
  const bf16* p = base + m * ld + 8 * h;
  bf16x8 lo = *(const bf16x8*)p;
  bf16x8 hi = *(const bf16x8*)(p + 16);
  return __builtin_shufflevector(lo, hi, 0,1,2,3,4,5,6,7,8,9,10,11,12,13,14,15);
}

// B operand (32x16 bf16, KxN): lane (n = lane&15, h = lane>>4) holds column n,
// VGPR0-3 = K[8h .. 8h+7], VGPR4-7 = K[16+8h .. 16+8h+7]. Column n of B == row n of B^T, so pass
// B^T row-major: elements 8h..8h+7 and 16+8h..16+8h+7 of row n.
__device__ __forceinline__ bf16x16 load_b(const bf16* baseT, int ld) {
  int lane = threadIdx.x & 31;
  int n = lane & 15, h = lane >> 4;
  const bf16* p = baseT + n * ld + 8 * h;
  bf16x8 lo = *(const bf16x8*)p;
  bf16x8 hi = *(const bf16x8*)(p + 16);
  return __builtin_shufflevector(lo, hi, 0,1,2,3,4,5,6,7,8,9,10,11,12,13,14,15);
}

__device__ __forceinline__ v8f wmma_bf16(bf16x16 a, bf16x16 b, v8f c) {
  return __builtin_amdgcn_wmma_f32_16x16x32_bf16(false, a, false, b, (short)0, c, false, false);
}

// ---------- CDNA5 async global->LDS (ASYNCcnt path) ----------
// ROCm signature (revealed by diagnostics): param1 = v4i in addrspace(1),
// gcc-style vector_size(16) int vector; param2 = LDS-side pointer; 2 imms.
typedef int v4i_vs __attribute__((vector_size(4 * sizeof(int))));
typedef __attribute__((address_space(1))) v4i_vs* v4i_gptr;
typedef __attribute__((address_space(3))) v4i_vs* v4i_lptr;

__device__ __forceinline__ void async_copy16(const bf16* g, bf16* l) {
#if __has_builtin(__builtin_amdgcn_global_load_async_to_lds_b128)
  __builtin_amdgcn_global_load_async_to_lds_b128(
      (v4i_gptr)(void*)g, (v4i_lptr)(void*)l, 0, 0);
#else
  *(bf16x8*)l = *(const bf16x8*)g;
#endif
}
__device__ __forceinline__ void wait_async0() {
#if __has_builtin(__builtin_amdgcn_s_wait_asynccnt)
  __builtin_amdgcn_s_wait_asynccnt(0);
#else
  asm volatile("s_wait_asynccnt 0x0" ::: "memory");
#endif
}

// stage one 32-key tile: K via async global->LDS, V via transpose through VGPRs
__device__ __forceinline__ void stage_tiles(const bf16* __restrict__ Kb,
                                            const bf16* __restrict__ Vb,
                                            int k0, bf16* sKbuf, bf16* sVtbuf,
                                            int tid) {
  // K tile (32 x 256 bf16 = 16KB) : 4 async 16B chunks per thread
#pragma unroll
  for (int c = 0; c < 4; ++c) {
    int i = (tid + c * 256) * 8;
    async_copy16(Kb + (size_t)k0 * DMODEL + i, sKbuf + i);
  }
  // V tile transposed: sVt[d][j] = V[k0+j][d]
  int j = tid >> 3;
  int dbase = (tid & 7) * 32;
  const bf16* vr = Vb + (size_t)(k0 + j) * DMODEL + dbase;
#pragma unroll
  for (int u = 0; u < 4; ++u) {
    bf16x8 x = *(const bf16x8*)(vr + u * 8);
#pragma unroll
    for (int e = 0; e < 8; ++e)
      sVtbuf[(dbase + u * 8 + e) * 32 + j] = x[e];
  }
}

// ---------- kernel A (placed FIRST so the disasm window shows its loop):
// flash attention + residual + layernorm.
// grid: B * (S/128) blocks of 256 thr. wave owns 16 query rows, full D=256 O.
// Q operand register-resident (qA[8] = 64 VGPRs); K async-staged, double-buffered.
__global__ void __launch_bounds__(256, 1)
flash_kernel(const bf16* __restrict__ Q,
             const bf16* __restrict__ K,
             const bf16* __restrict__ V,
             const float* __restrict__ resid,  // transformer_embedding
             const float* __restrict__ gamma,
             const float* __restrict__ beta,
             float* __restrict__ out) {
  extern __shared__ __attribute__((aligned(16))) bf16 lds[];
  bf16* sK0  = lds;                  // 32*256 (16KB)
  bf16* sK1  = sK0 + 32 * DMODEL;    // 32*256 (16KB)
  bf16* sVt0 = sK1 + 32 * DMODEL;    // 256*32 (16KB)
  bf16* sVt1 = sVt0 + DMODEL * 32;   // 256*32 (16KB)
  bf16* sP   = sVt1 + DMODEL * 32;   // 8*16*32 (8KB), per-wave P tile

  int tid = threadIdx.x, lane = tid & 31, wave = tid >> 5;
  int b  = blockIdx.x >> 5;        // 32 q-tiles per batch
  int qt = blockIdx.x & 31;
  int q0 = qt * 128;

  const bf16* Kb = K + (size_t)b * 4096 * DMODEL;
  const bf16* Vb = V + (size_t)b * 4096 * DMODEL;

  int m0 = wave * 16;
  int col = lane & 15, h = lane >> 4;

  // register-resident Q operand for this wave's 16 rows (pre-scaled)
  bf16x16 qA[8];
  {
    const bf16* qrow = Q + ((size_t)b * 4096 + q0 + m0) * DMODEL;
#pragma unroll
    for (int ks = 0; ks < 8; ++ks)
      qA[ks] = load_a(qrow + ks * 32, DMODEL);
  }

  v8f acc[16];
#pragma unroll
  for (int t = 0; t < 16; ++t)
    acc[t] = (v8f){0.f,0.f,0.f,0.f,0.f,0.f,0.f,0.f};
  float mrow[8], lrow[8];
#pragma unroll
  for (int v = 0; v < 8; ++v) { mrow[v] = -3.0e38f; lrow[v] = 0.f; }

  // prologue: stage tile 0 into buffer 0
  stage_tiles(Kb, Vb, 0, sK0, sVt0, tid);
  wait_async0();
  __syncthreads();

  for (int kt = 0; kt < 128; ++kt) {
    int cur = kt & 1;
    bf16* sKc  = cur ? sK1 : sK0;
    bf16* sVtc = cur ? sVt1 : sVt0;

    // launch next tile into the other buffer while computing on this one
    if (kt < 127)
      stage_tiles(Kb, Vb, (kt + 1) * 32, cur ? sK0 : sK1, cur ? sVt0 : sVt1, tid);
    if (kt < 126) {
      __builtin_prefetch(Kb + (size_t)(kt + 2) * 32 * DMODEL + tid * 32, 0, 3);
      __builtin_prefetch(Vb + (size_t)(kt + 2) * 32 * DMODEL + tid * 32, 0, 3);
    }

    // S = Q K^T  (two 16x16 n-tiles: keys 0-15 and 16-31)
    v8f s0 = {0.f,0.f,0.f,0.f,0.f,0.f,0.f,0.f};
    v8f s1 = {0.f,0.f,0.f,0.f,0.f,0.f,0.f,0.f};
#pragma unroll
    for (int ks = 0; ks < 8; ++ks) {
      s0 = wmma_bf16(qA[ks], load_b(sKc + ks * 32, DMODEL), s0);
      s1 = wmma_bf16(qA[ks], load_b(sKc + 16 * DMODEL + ks * 32, DMODEL), s1);
    }

    // online softmax: C-layout row of VGPR v is (v + 8h), cols = lane group
    float alpha[8];
#pragma unroll
    for (int v = 0; v < 8; ++v) {
      float x0 = s0[v];
      float x1 = s1[v];
      float rm = red_max16(fmaxf(x0, x1));
      float mnew = fmaxf(mrow[v], rm);
      float p0 = __expf(x0 - mnew);
      float p1 = __expf(x1 - mnew);
      float rs = red_sum16(p0 + p1);
      alpha[v] = __expf(mrow[v] - mnew);
      lrow[v] = lrow[v] * alpha[v] + rs;
      mrow[v] = mnew;
      bf16* pb = sP + wave * 512 + (v + 8 * h) * 32 + col;  // P row-major 16x32
      pb[0]  = (bf16)p0;
      pb[16] = (bf16)p1;
    }
    asm volatile("s_wait_dscnt 0" ::: "memory");  // P stores visible to own wave

    // O = diag(alpha) O + P V
    bf16x16 aP = load_a(sP + wave * 512, 32);
#pragma unroll
    for (int t = 0; t < 16; ++t) {
#pragma unroll
      for (int v = 0; v < 8; ++v) acc[t][v] *= alpha[v];
      acc[t] = wmma_bf16(aP, load_b(sVtc + t * 16 * 32, 32), acc[t]);
    }

    wait_async0();    // next-tile async K writes have landed
    __syncthreads();  // publish next buffer / retire reads of current
  }

  // epilogue: O/l + residual, then LayerNorm over D=256 (in-wave)
  int rbase = b * 4096 + q0 + m0 + 8 * h;
  float invl[8];
#pragma unroll
  for (int v = 0; v < 8; ++v) invl[v] = 1.0f / lrow[v];
  float g[16], be[16];
#pragma unroll
  for (int t = 0; t < 16; ++t) {
    g[t]  = gamma[t * 16 + col];
    be[t] = beta[t * 16 + col];
  }
#pragma unroll
  for (int v = 0; v < 8; ++v) {
    int row = rbase + v;
    float vals[16];
    float sum = 0.f;
#pragma unroll
    for (int t = 0; t < 16; ++t) {
      float o = acc[t][v] * invl[v] + resid[(size_t)row * DMODEL + t * 16 + col];
      vals[t] = o;
      sum += o;
    }
    sum = red_sum16(sum);
    float mu = sum * (1.0f / 256.0f);
    float s2 = 0.f;
#pragma unroll
    for (int t = 0; t < 16; ++t) {
      float d = vals[t] - mu;
      s2 += d * d;
    }
    s2 = red_sum16(s2);
    float rstd = rsqrtf(s2 * (1.0f / 256.0f) + EPS);
#pragma unroll
    for (int t = 0; t < 16; ++t)
      out[(size_t)row * DMODEL + t * 16 + col] = (vals[t] - mu) * rstd * g[t] + be[t];
  }
}

// ---------- kernel B: weight fp32 -> bf16 ----------
__global__ void cvt_w_kernel(const float* __restrict__ wq,
                             const float* __restrict__ wk,
                             const float* __restrict__ wv,
                             bf16* __restrict__ o) {
  int i = blockIdx.x * blockDim.x + threadIdx.x;
  if (i < DMODEL * DMODEL) {
    o[i]                       = (bf16)wq[i];
    o[DMODEL * DMODEL + i]     = (bf16)wk[i];
    o[2 * DMODEL * DMODEL + i] = (bf16)wv[i];
  }
}

// ---------- kernel C: QKV projection, bf16 out ----------
// block: 256 thr (8 waves), 128 rows. wave w computes rows 16w..16w+15.
// Q is pre-scaled by 1/sqrt(D) so the flash kernel's scores need no scaling.
__global__ void __launch_bounds__(256, 1)
qkv_kernel(const float* __restrict__ xs,  // supervised  (K,V input)
           const float* __restrict__ xt,  // transformer (Q input)
           const float* __restrict__ bq,
           const float* __restrict__ bk,
           const float* __restrict__ bv,
           const bf16* __restrict__ wq,
           const bf16* __restrict__ wk,
           const bf16* __restrict__ wv,
           bf16* __restrict__ Q,
           bf16* __restrict__ K,
           bf16* __restrict__ V) {
  extern __shared__ __attribute__((aligned(16))) bf16 lds[];
  bf16* sXT = lds;                   // 128*256 bf16 (64KB)
  bf16* sXS = lds + 128 * DMODEL;    // 128*256 bf16 (64KB)

  int tid = threadIdx.x;
  int row0 = blockIdx.x * 128;

  const float4* xt4 = (const float4*)(xt + (size_t)row0 * DMODEL);
  const float4* xs4 = (const float4*)(xs + (size_t)row0 * DMODEL);
  for (int i = tid; i < 128 * DMODEL / 4; i += 256) {
    float4 a = xt4[i];
    float4 b = xs4[i];
    bf16x4 ab = {(bf16)a.x, (bf16)a.y, (bf16)a.z, (bf16)a.w};
    bf16x4 bb = {(bf16)b.x, (bf16)b.y, (bf16)b.z, (bf16)b.w};
    *(bf16x4*)(sXT + i * 4) = ab;
    *(bf16x4*)(sXS + i * 4) = bb;
  }
  __syncthreads();

  int wave = tid >> 5, lane = tid & 31;
  int m0 = wave * 16;
  int col = lane & 15, h = lane >> 4;
  int rbase = row0 + m0 + 8 * h;   // + v = output row of acc VGPR v

  bf16x16 aT[8];

  // ---- Pass 1: Q = (xt @ Wq^T + bq) * (1/sqrt(D)) ----
#pragma unroll
  for (int ks = 0; ks < 8; ++ks)
    aT[ks] = load_a(sXT + m0 * DMODEL + ks * 32, DMODEL);
  for (int nt = 0; nt < 16; ++nt) {
    int ncol = nt * 16 + col;
    v8f acc = {0.f,0.f,0.f,0.f,0.f,0.f,0.f,0.f};
#pragma unroll
    for (int ks = 0; ks < 8; ++ks)
      acc = wmma_bf16(aT[ks], load_b(wq + nt * 16 * DMODEL + ks * 32, DMODEL), acc);
    float bias = bq[ncol];
#pragma unroll
    for (int v = 0; v < 8; ++v)
      Q[(size_t)(rbase + v) * DMODEL + ncol] = (bf16)((acc[v] + bias) * 0.0625f);
  }

  // ---- Pass 2: K and V share the same A operand (xs) ----
#pragma unroll
  for (int ks = 0; ks < 8; ++ks)
    aT[ks] = load_a(sXS + m0 * DMODEL + ks * 32, DMODEL);
  for (int nt = 0; nt < 16; ++nt) {
    int ncol = nt * 16 + col;
    v8f accK = {0.f,0.f,0.f,0.f,0.f,0.f,0.f,0.f};
    v8f accV = {0.f,0.f,0.f,0.f,0.f,0.f,0.f,0.f};
#pragma unroll
    for (int ks = 0; ks < 8; ++ks) {
      accK = wmma_bf16(aT[ks], load_b(wk + nt * 16 * DMODEL + ks * 32, DMODEL), accK);
      accV = wmma_bf16(aT[ks], load_b(wv + nt * 16 * DMODEL + ks * 32, DMODEL), accV);
    }
    float biasK = bk[ncol];
    float biasV = bv[ncol];
#pragma unroll
    for (int v = 0; v < 8; ++v) {
      K[(size_t)(rbase + v) * DMODEL + ncol] = (bf16)(accK[v] + biasK);
      V[(size_t)(rbase + v) * DMODEL + ncol] = (bf16)(accV[v] + biasV);
    }
  }
}

// ---------- launcher ----------
extern "C" void kernel_launch(void* const* d_in, const int* in_sizes, int n_in,
                              void* d_out, int out_size, void* d_ws, size_t ws_size,
                              hipStream_t stream) {
  (void)in_sizes; (void)n_in; (void)out_size; (void)ws_size;
  const float* sup   = (const float*)d_in[0];
  const float* tra   = (const float*)d_in[1];
  const float* Wq    = (const float*)d_in[2];
  const float* bq    = (const float*)d_in[3];
  const float* Wk    = (const float*)d_in[4];
  const float* bk    = (const float*)d_in[5];
  const float* Wv    = (const float*)d_in[6];
  const float* bv    = (const float*)d_in[7];
  const float* gamma = (const float*)d_in[8];
  const float* beta  = (const float*)d_in[9];
  float* out = (float*)d_out;

  bf16* ws  = (bf16*)d_ws;
  bf16* wqb = ws;
  bf16* wkb = ws + DMODEL * DMODEL;
  bf16* wvb = ws + 2 * DMODEL * DMODEL;
  bf16* Qb  = ws + 3 * DMODEL * DMODEL;
  bf16* Kb  = Qb + (size_t)8 * 4096 * DMODEL;
  bf16* Vb  = Kb + (size_t)8 * 4096 * DMODEL;

  cvt_w_kernel<<<DMODEL * DMODEL / 256, 256, 0, stream>>>(Wq, Wk, Wv, ws);

  size_t lds_qkv = (size_t)2 * 128 * DMODEL * sizeof(bf16);          // 128 KB
  qkv_kernel<<<8 * 4096 / 128, 256, lds_qkv, stream>>>(
      sup, tra, bq, bk, bv, wqb, wkb, wvb, Qb, Kb, Vb);

  size_t lds_fa = (size_t)(4 * 32 * DMODEL + 8 * 16 * 32) * sizeof(bf16);  // 72 KB
  flash_kernel<<<8 * 4096 / 128, 256, lds_fa, stream>>>(
      Qb, Kb, Vb, tra, gamma, beta, out);
}
